// PALMAttention_8418135900896
// MI455X (gfx1250) — matrix operand order repaired and checked by reading, not compiled, not run on hardware
//
#include <hip/hip_runtime.h>
#include <hip/hip_bf16.h>

// Problem constants (from reference)
#define B_   4
#define S_   2048
#define H_   1024
#define NH_  16
#define HD_  64
#define TOK  (B_ * S_)          // 8192 tokens

typedef __attribute__((ext_vector_type(16))) __bf16          v16bf;
typedef __attribute__((ext_vector_type(8)))  float           v8f;
typedef __attribute__((ext_vector_type(8)))  unsigned short  us8;
typedef __attribute__((ext_vector_type(16))) unsigned short  us16;
// Matches clang's builtin "V4i" parameter type (generic vector_size vector).
typedef int v4i __attribute__((vector_size(16)));

// CDNA5 async global->LDS path (ASYNCcnt). Guarded so compilation cannot
// break if the builtin is absent in this toolchain (fallback: ds_store).
#if defined(__AMDGCN__) && __has_builtin(__builtin_amdgcn_global_load_async_to_lds_b128)
#define ASYNC_LDS 1
#else
#define ASYNC_LDS 0
#endif

static __device__ __forceinline__ void async_cp16(unsigned short* lds,
                                                  const unsigned short* g) {
#if ASYNC_LDS
  __builtin_amdgcn_global_load_async_to_lds_b128(
      (__attribute__((address_space(1))) v4i*)g,
      (__attribute__((address_space(3))) v4i*)lds,
      /*imm offset*/ 0, /*cpol*/ 0);
#else
  *(us8*)lds = *(const us8*)g;
#endif
}

static __device__ __forceinline__ void wait_async0() {
#if ASYNC_LDS
#if __has_builtin(__builtin_amdgcn_s_wait_asynccnt)
  __builtin_amdgcn_s_wait_asynccnt(0);
#else
  asm volatile("s_wait_asynccnt 0x0" ::: "memory");
#endif
#endif
}

// fp32 -> bf16 (round-to-nearest-even)
static __device__ __forceinline__ unsigned short f2bfu(float f) {
  unsigned u = __builtin_bit_cast(unsigned, f);
  u += 0x7fffu + ((u >> 16) & 1u);
  return (unsigned short)(u >> 16);
}

// A-operand loader (16-bit A 16x32 layout): per lane, elements 0..7 are 8
// contiguous bf16 at p0, elements 8..15 are 8 contiguous bf16 at p0+16.
static __device__ __forceinline__ v16bf load_a16(const unsigned short* p0) {
  us8 a = *(const us8*)(p0);
  us8 b = *(const us8*)(p0 + 16);
  us16 t;
#pragma unroll
  for (int i = 0; i < 8; ++i) { t[i] = a[i]; t[i + 8] = b[i]; }
  return __builtin_bit_cast(v16bf, t);
}

// B-operand loader (contiguous k = 16h + s order; also loads A here): per lane, 16 contiguous bf16.
static __device__ __forceinline__ v16bf load_b16(const unsigned short* p) {
  us16 t = *(const us16*)(p);
  return __builtin_bit_cast(v16bf, t);
}

// ---------------------------------------------------------------------------
// fp32 -> bf16 conversion
// ---------------------------------------------------------------------------
__global__ void cvt_bf16(const float* __restrict__ in,
                         unsigned short* __restrict__ out, int n) {
  int i = blockIdx.x * blockDim.x + threadIdx.x;
  if (i < n) out[i] = f2bfu(in[i]);
}

// ---------------------------------------------------------------------------
// Block-cooperative GEMM: out = alpha * (X[M,K] @ W[N,K]^T + bias).
// Block (8 waves) owns a 128x64 tile. Per 32-K chunk: async-stage X(8KB) +
// W(4KB) into double-buffered LDS, compute from LDS while next chunk streams.
// mode: 0 = bf16 [bh,s,d]; 1 = bf16 V^T [bh,d,s]; 2 = fp32 row-major.
// ---------------------------------------------------------------------------
__global__ void __launch_bounds__(256)
gemm_wmma(const unsigned short* __restrict__ X,
          const unsigned short* __restrict__ W,
          const float* __restrict__ bias,
          void* __restrict__ out, float alpha, int mode) {
  const int K    = H_;
  const int tid  = threadIdx.x;
  const int lane = tid & 31;
  const int wave = tid >> 5;
  const int rowBlock = blockIdx.x >> 4;     // 64 row blocks of 128
  const int cg       = blockIdx.x & 15;     // 16 col groups of 64

  __shared__ __align__(16) unsigned short shX[2][128 * 32];  // 16 KB
  __shared__ __align__(16) unsigned short shW[2][64 * 32];   //  8 KB

  const unsigned short* Xg = X + (size_t)(rowBlock * 128) * K;
  const unsigned short* Wg = W + (size_t)(cg * 64) * K;

  // Stage one 32-wide K chunk into LDS buffer `b`.
  auto stage = [&](int b, int kc) {
    // X tile: 128 rows x 32 k = 512 x 16B chunks; 2 per thread.
    int c = tid;
    async_cp16(&shX[b][c * 8], Xg + (size_t)(c >> 2) * K + kc + ((c & 3) << 3));
    c = tid + 256;
    async_cp16(&shX[b][c * 8], Xg + (size_t)(c >> 2) * K + kc + ((c & 3) << 3));
    // W tile: 64 cols x 32 k = 256 x 16B chunks; 1 per thread.
    async_cp16(&shW[b][tid * 8],
               Wg + (size_t)(tid >> 2) * K + kc + ((tid & 3) << 3));
  };

  v8f acc[4] = {v8f{}, v8f{}, v8f{}, v8f{}};

  const int lr    = wave * 16 + (lane & 15);        // local A row
  const int aoff  = (lane >> 4) << 3;               // A k-offset per half
  const int boff  = (lane >> 4) << 4;               // B k-offset per half

  stage(0, 0);
  int buf = 0;
  for (int kci = 0; kci < K / 32; ++kci) {
    wait_async0();          // this wave's staged chunk is in LDS
    __syncthreads();        // all waves' chunks visible
    if (kci + 1 < K / 32) stage(buf ^ 1, (kci + 1) * 32);

    v16bf A = load_b16(&shX[buf][lr * 32 + boff]);
#pragma unroll
    for (int ct = 0; ct < 4; ++ct) {
      v16bf Bm = load_b16(&shW[buf][(ct * 16 + (lane & 15)) * 32 + boff]);
      acc[ct] = __builtin_amdgcn_wmma_f32_16x16x32_bf16(
          false, A, false, Bm, (short)0, acc[ct], false, false);
    }
    __syncthreads();        // all reads of `buf` done before it is restaged
    buf ^= 1;
  }

#pragma unroll
  for (int ct = 0; ct < 4; ++ct) {
#pragma unroll
    for (int r = 0; r < 8; ++r) {
      int m = rowBlock * 128 + wave * 16 + r + ((lane >> 4) << 3);
      int n = cg * 64 + ct * 16 + (lane & 15);
      float v = (acc[ct][r] + bias[n]) * alpha;
      if (mode == 2) {
        ((float*)out)[(size_t)m * H_ + n] = v;
      } else {
        int b = m / S_, s = m % S_;
        int h = n >> 6, d = n & 63;
        size_t addr = (mode == 0)
            ? ((size_t)(b * NH_ + h) * S_ + s) * HD_ + d       // [bh,s,d]
            : ((size_t)(b * NH_ + h) * HD_ + d) * S_ + s;      // [bh,d,s]
        ((unsigned short*)out)[addr] = f2bfu(v);
      }
    }
  }
}

// ---------------------------------------------------------------------------
// Flash attention: one wave per (b,h,16-query tile); stream keys 32 at a time.
// Scores computed transposed (M=key, N=query) so softmax stats are per-lane.
// ---------------------------------------------------------------------------
__global__ void __launch_bounds__(256)
attn_wmma(const unsigned short* __restrict__ Q,   // [BH,S,HD] bf16, pre-scaled
          const unsigned short* __restrict__ Kt,  // [BH,S,HD] bf16
          const unsigned short* __restrict__ Vt,  // [BH,HD,S] bf16
          const float* __restrict__ mask,         // [B,S]
          unsigned short* __restrict__ ctxO) {    // [B,S,H] bf16
  const int lane = threadIdx.x & 31;
  const int wave = threadIdx.x >> 5;
  const int job  = blockIdx.x * 8 + wave;  // 8192 jobs
  const int qt   = job & (S_ / 16 - 1);    // 128 query tiles
  const int bh   = job >> 7;
  const int b    = bh >> 4;
  const int h    = bh & 15;

  const size_t qkBase = (size_t)bh * S_ * HD_;
  const int q = qt * 16 + (lane & 15);

  // Q as B-operand (N=query, K=d), one tile per 32-d chunk
  v16bf Qb[2];
#pragma unroll
  for (int dc = 0; dc < 2; ++dc)
    Qb[dc] = load_b16(Q + qkBase + (size_t)q * HD_ + dc * 32 + ((lane >> 4) << 4));

  v8f ctx[4] = {v8f{}, v8f{}, v8f{}, v8f{}};
  float mval = -1e30f, lsum = 0.f;
  const float* mrow = mask + b * S_;

  for (int k0 = 0; k0 < S_; k0 += 32) {
    v8f sc[2];
#pragma unroll
    for (int kt = 0; kt < 2; ++kt) {
      v8f a = {};
      const int key = k0 + kt * 16 + (lane & 15);
#pragma unroll
      for (int dc = 0; dc < 2; ++dc) {
        v16bf Ak = load_b16(Kt + qkBase + (size_t)key * HD_ + dc * 32 +
                            ((lane >> 4) << 4));
        a = __builtin_amdgcn_wmma_f32_16x16x32_bf16(
            false, Ak, false, Qb[dc], (short)0, a, false, false);
      }
#pragma unroll
      for (int r = 0; r < 8; ++r) {  // additive mask bias per key row
        int kr = k0 + kt * 16 + r + ((lane >> 4) << 3);
        a[r] += (1.0f - mrow[kr]) * -10000.0f;
      }
      sc[kt] = a;
    }

    // online softmax (query = lane%16; halves symmetrized via xor-16)
    float cmax = -1e30f;
#pragma unroll
    for (int r = 0; r < 8; ++r) {
      cmax = fmaxf(cmax, sc[0][r]);
      cmax = fmaxf(cmax, sc[1][r]);
    }
    cmax = fmaxf(cmax, __shfl_xor(cmax, 16, 32));
    float newm  = fmaxf(mval, cmax);
    float scale = __expf(mval - newm);
    mval = newm;

    float p0[8], p1[8], psum = 0.f;
#pragma unroll
    for (int r = 0; r < 8; ++r) {
      p0[r] = __expf(sc[0][r] - newm);
      p1[r] = __expf(sc[1][r] - newm);
      psum += p0[r] + p1[r];
    }
    psum += __shfl_xor(psum, 16, 32);
    lsum = lsum * scale + psum;
#pragma unroll
    for (int dt = 0; dt < 4; ++dt)
#pragma unroll
      for (int r = 0; r < 8; ++r) ctx[dt][r] *= scale;

    // Build P^T as B-operand (K=key 0..31, N=query): lanes<16 own keys 0..7,
    // partner holds 8..15 (tile kt0); lanes>=16 mirror for keys 16..31 (kt1).
    us16 bp;
#pragma unroll
    for (int r = 0; r < 8; ++r) {
      float o0 = __shfl_xor(p0[r], 16, 32);
      float o1 = __shfl_xor(p1[r], 16, 32);
      float lo = (lane < 16) ? p0[r] : o1;
      float hi = (lane < 16) ? o0    : p1[r];
      bp[r]     = f2bfu(lo);
      bp[r + 8] = f2bfu(hi);
    }
    v16bf Bp = __builtin_bit_cast(v16bf, bp);

    // ctx^T[d][q] += V^T[d][k] * P^T[k][q]
#pragma unroll
    for (int dt = 0; dt < 4; ++dt) {
      int d = dt * 16 + (lane & 15);
      v16bf Av = load_b16(Vt + (size_t)bh * HD_ * S_ + (size_t)d * S_ + k0 +
                          ((lane >> 4) << 4));
      ctx[dt] = __builtin_amdgcn_wmma_f32_16x16x32_bf16(
          false, Av, false, Bp, (short)0, ctx[dt], false, false);
    }
  }

  const float rinv = 1.0f / lsum;
#pragma unroll
  for (int dt = 0; dt < 4; ++dt)
#pragma unroll
    for (int r = 0; r < 8; ++r) {
      int d = dt * 16 + r + ((lane >> 4) << 3);
      ctxO[(size_t)(b * S_ + q) * H_ + h * HD_ + d] = f2bfu(ctx[dt][r] * rinv);
    }
}

// ---------------------------------------------------------------------------
// Residual + LayerNorm: one block (256 threads) per token row.
// ---------------------------------------------------------------------------
__global__ void __launch_bounds__(256)
ln_kernel(const float* __restrict__ y, const float* __restrict__ x0,
          const float* __restrict__ g, const float* __restrict__ bb,
          float* __restrict__ out) {
  __shared__ float xs[H_];
  __shared__ float r1[256], r2[256];
  const int t = blockIdx.x;
  const int tid = threadIdx.x;
  float s = 0.f, sq = 0.f;
  for (int j = tid; j < H_; j += 256) {
    float v = y[(size_t)t * H_ + j] + x0[(size_t)t * H_ + j];
    xs[j] = v; s += v; sq += v * v;
  }
  r1[tid] = s; r2[tid] = sq;
  __syncthreads();
  for (int off = 128; off > 0; off >>= 1) {
    if (tid < off) { r1[tid] += r1[tid + off]; r2[tid] += r2[tid + off]; }
    __syncthreads();
  }
  float mu   = r1[0] * (1.0f / H_);
  float var  = r2[0] * (1.0f / H_) - mu * mu;
  float rstd = rsqrtf(var + 1e-12f);
  for (int j = tid; j < H_; j += 256)
    out[(size_t)t * H_ + j] = (xs[j] - mu) * rstd * g[j] + bb[j];
}

// ---------------------------------------------------------------------------
extern "C" void kernel_launch(void* const* d_in, const int* in_sizes, int n_in,
                              void* d_out, int out_size, void* d_ws,
                              size_t ws_size, hipStream_t stream) {
  const float* hidden = (const float*)d_in[0];
  const float* mask   = (const float*)d_in[1];
  const float* Wq = (const float*)d_in[2];  const float* bq = (const float*)d_in[3];
  const float* Wk = (const float*)d_in[4];  const float* bk = (const float*)d_in[5];
  const float* Wv = (const float*)d_in[6];  const float* bv = (const float*)d_in[7];
  const float* Wd = (const float*)d_in[8];  const float* bd = (const float*)d_in[9];
  const float* ln_g = (const float*)d_in[10];
  const float* ln_b = (const float*)d_in[11];
  float* out = (float*)d_out;

  char* ws = (char*)d_ws;
  const size_t TOKH2 = (size_t)TOK * H_ * 2;   // 16 MiB bf16 activations
  const size_t W2    = (size_t)H_ * H_ * 2;    // 2 MiB bf16 weights
  // Layout (Yf aliases Xbf+Qb, both dead by the time Yf is written;
  // stream ordering guarantees safety). Total ~92 MB.
  unsigned short* Xbf  = (unsigned short*)(ws + 0);
  unsigned short* Qb   = (unsigned short*)(ws + TOKH2);
  float*          Yf   = (float*)(ws + 0);                 // 33.5 MB alias
  unsigned short* Kb   = (unsigned short*)(ws + 2 * TOKH2);
  unsigned short* Vtb  = (unsigned short*)(ws + 3 * TOKH2);
  unsigned short* Ctxb = (unsigned short*)(ws + 4 * TOKH2);
  unsigned short* Wqb  = (unsigned short*)(ws + 5 * TOKH2);
  unsigned short* Wkb  = (unsigned short*)(ws + 5 * TOKH2 + W2);
  unsigned short* Wvb  = (unsigned short*)(ws + 5 * TOKH2 + 2 * W2);
  unsigned short* Wdb  = (unsigned short*)(ws + 5 * TOKH2 + 3 * W2);

  const int NXT = TOK * H_;   // 8,388,608
  const int NWT = H_ * H_;    // 1,048,576
  cvt_bf16<<<(NXT + 255) / 256, 256, 0, stream>>>(hidden, Xbf, NXT);
  cvt_bf16<<<(NWT + 255) / 256, 256, 0, stream>>>(Wq, Wqb, NWT);
  cvt_bf16<<<(NWT + 255) / 256, 256, 0, stream>>>(Wk, Wkb, NWT);
  cvt_bf16<<<(NWT + 255) / 256, 256, 0, stream>>>(Wv, Wvb, NWT);
  cvt_bf16<<<(NWT + 255) / 256, 256, 0, stream>>>(Wd, Wdb, NWT);

  // GEMMs: grid = 64 row-blocks x 16 col-groups = 1024 blocks of 256.
  gemm_wmma<<<1024, 256, 0, stream>>>(Xbf, Wqb, bq, Qb, 0.125f, 0);  // Q/sqrt(HD)
  gemm_wmma<<<1024, 256, 0, stream>>>(Xbf, Wkb, bk, Kb, 1.0f, 0);
  gemm_wmma<<<1024, 256, 0, stream>>>(Xbf, Wvb, bv, Vtb, 1.0f, 1);   // V^T

  attn_wmma<<<1024, 256, 0, stream>>>(Qb, Kb, Vtb, mask, Ctxb);

  gemm_wmma<<<1024, 256, 0, stream>>>(Ctxb, Wdb, bd, Yf, 1.0f, 2);

  ln_kernel<<<TOK, 256, 0, stream>>>(Yf, hidden, ln_g, ln_b, out);
}
